// MOE_65592740545204
// MI455X (gfx1250) — matrix operand order repaired and checked by reading, not compiled, not run on hardware
//
#include <hip/hip_runtime.h>
#include <math.h>

// ---------------------------------------------------------------------------
// Types for CDNA5 WMMA (wave32): A 16x32 bf16, B 32x16 bf16, C/D 16x16 f32
// ---------------------------------------------------------------------------
typedef __attribute__((ext_vector_type(16))) __bf16 v16bf;
typedef __attribute__((ext_vector_type(8)))  __bf16 v8bf;
typedef __attribute__((ext_vector_type(4)))  __bf16 v4bf;
typedef __attribute__((ext_vector_type(8)))  float  v8f;
typedef __attribute__((ext_vector_type(4)))  float  v4f;

__device__ __forceinline__ __bf16 f2bf(float f) {
    union { float f; unsigned u; } v; v.f = f;
    unsigned r = (v.u + 0x7FFFu + ((v.u >> 16) & 1u)) >> 16;   // RNE
    unsigned short h = (unsigned short)r;
    __bf16 o; __builtin_memcpy(&o, &h, 2); return o;
}

__device__ __forceinline__ void atomAddF(float* p, float v) {
    unsafeAtomicAdd(p, v);   // native global_atomic_add_f32 on CDNA5
}

// ---------------------------------------------------------------------------
// GEMM: C[nrows,ncols] = A[nrows,kdim] @ B[kdim,ncols] (+bias), f32 in/out,
// bf16 WMMA compute. Block = 256 thr = 8 waves; block tile 64x64, K-step 32.
// Each wave owns a 16x32 C tile (one A fragment, two B fragments, 2 WMMA).
// ---------------------------------------------------------------------------
#define BM 64
#define BN 64
#define BK 32
#define LDP 40   // padded LDS K-stride (16B-aligned rows, spreads banks)

__global__ __launch_bounds__(256)
void gemm_bf16_wmma(const float* __restrict__ A, int lda,
                    const float* __restrict__ B, int ldb,
                    const float* __restrict__ bias,
                    float* __restrict__ C, int ldc,
                    int nrows, int kdim, int ncols)
{
    __shared__ alignas(16) __bf16 As[BM * LDP];   // [row][k]
    __shared__ alignas(16) __bf16 Bs[BN * LDP];   // [col][k]  (transposed)

    const int tid  = threadIdx.x;
    const int lane = tid & 31;
    const int wid  = tid >> 5;        // 0..7
    const int wr   = wid >> 1;        // wave row 0..3  (16 rows each)
    const int wc   = wid & 1;         // wave col 0..1  (32 cols each)
    const int h    = lane >> 4;       // lane half (K interleave)
    const int l15  = lane & 15;

    const int rowBase = blockIdx.y * BM;
    const int colBase = blockIdx.x * BN;
    const bool rowsFull = (rowBase + BM) <= nrows;
    const bool colsFull = (colBase + BN) <= ncols;

    v8f acc0 = {0.f,0.f,0.f,0.f,0.f,0.f,0.f,0.f};
    v8f acc1 = {0.f,0.f,0.f,0.f,0.f,0.f,0.f,0.f};

    for (int k0 = 0; k0 < kdim; k0 += BK) {
        const bool kFull = (k0 + BK) <= kdim;

        // ---- stage A tile (BM x BK), f32 -> bf16
        if (rowsFull && kFull) {                    // uniform fast path
            #pragma unroll
            for (int i = 0; i < (BM * BK) / (256 * 4); ++i) {
                int q  = tid + i * 256;             // float4 index
                int r  = q >> 3;                    // 8 float4 per row
                int c4 = (q & 7) * 4;
                v4f f = *(const v4f*)(A + (long)(rowBase + r) * lda + k0 + c4);
                v4bf o = { f2bf(f[0]), f2bf(f[1]), f2bf(f[2]), f2bf(f[3]) };
                *(v4bf*)&As[r * LDP + c4] = o;      // ds_store_b64
            }
        } else {
            #pragma unroll
            for (int i = 0; i < (BM * BK) / 256; ++i) {
                int e = tid + i * 256;
                int r = e >> 5, c = e & 31;
                int gr = rowBase + r, gk = k0 + c;
                float v = (gr < nrows && gk < kdim) ? A[(long)gr * lda + gk] : 0.f;
                As[r * LDP + c] = f2bf(v);
            }
        }

        // ---- stage B tile (BK x BN) transposed -> Bs[n][k]
        if (colsFull && kFull) {
            #pragma unroll
            for (int i = 0; i < (BK * BN) / (256 * 4); ++i) {
                int q   = tid + i * 256;
                int kk  = q >> 4;                   // 16 float4 per k-row
                int nn4 = (q & 15) * 4;
                v4f f = *(const v4f*)(B + (long)(k0 + kk) * ldb + colBase + nn4);
                #pragma unroll
                for (int j = 0; j < 4; ++j)
                    Bs[(nn4 + j) * LDP + kk] = f2bf(f[j]);
            }
        } else {
            #pragma unroll
            for (int i = 0; i < (BK * BN) / 256; ++i) {
                int e = tid + i * 256;
                int kk = e >> 6, nn = e & 63;
                int gk = k0 + kk, gn = colBase + nn;
                float v = (gk < kdim && gn < ncols) ? B[(long)gk * ldb + gn] : 0.f;
                Bs[nn * LDP + kk] = f2bf(v);
            }
        }
        __syncthreads();

        // ---- A fragment: elems 0..7 = K h*8.., 8..15 = K 16+h*8..
        const __bf16* ap = &As[(wr * 16 + l15) * LDP];
        v8bf alo = *(const v8bf*)(ap + h * 8);
        v8bf ahi = *(const v8bf*)(ap + 16 + h * 8);
        v16bf afrag;
        #pragma unroll
        for (int i = 0; i < 8; ++i) { afrag[i] = alo[i]; afrag[8 + i] = ahi[i]; }

        // ---- two B fragments / two WMMA per wave
        {
            const __bf16* bp = &Bs[(wc * 32 + l15) * LDP];
            v8bf blo = *(const v8bf*)(bp + h * 8);
            v8bf bhi = *(const v8bf*)(bp + 16 + h * 8);
            v16bf bfrag;
            #pragma unroll
            for (int i = 0; i < 8; ++i) { bfrag[i] = blo[i]; bfrag[8 + i] = bhi[i]; }
            acc0 = __builtin_amdgcn_wmma_f32_16x16x32_bf16(
                       false, afrag, false, bfrag, (short)0, acc0, false, false);
        }
        {
            const __bf16* bp = &Bs[(wc * 32 + 16 + l15) * LDP];
            v8bf blo = *(const v8bf*)(bp + h * 8);
            v8bf bhi = *(const v8bf*)(bp + 16 + h * 8);
            v16bf bfrag;
            #pragma unroll
            for (int i = 0; i < 8; ++i) { bfrag[i] = blo[i]; bfrag[8 + i] = bhi[i]; }
            acc1 = __builtin_amdgcn_wmma_f32_16x16x32_bf16(
                       false, afrag, false, bfrag, (short)0, acc1, false, false);
        }
        __syncthreads();
    }

    // ---- epilogue: acc[v] holds C[wr*16 + h*8 + v][...]
    #pragma unroll
    for (int t = 0; t < 2; ++t) {
        int cn = colBase + wc * 32 + t * 16 + l15;
        if (cn < ncols) {
            float bv = bias ? bias[cn] : 0.f;
            #pragma unroll
            for (int v = 0; v < 8; ++v) {
                int rm = rowBase + wr * 16 + h * 8 + v;
                float av = t ? acc1[v] : acc0[v];
                if (rm < nrows) C[(long)rm * ldc + cn] = av + bv;
            }
        }
    }
}

// ---------------------------------------------------------------------------
// Elementwise / sparse kernels
// ---------------------------------------------------------------------------
__global__ void zero_f32(float* p, size_t n) {
    size_t i = (size_t)blockIdx.x * blockDim.x + threadIdx.x;
    if (i < n) p[i] = 0.f;
}

// one wave32 per edge: Y[row] += val * X[col] (optionally X^2); vector gather
template<int D, bool SQ>
__global__ __launch_bounds__(256)
void spmm_scatter(const float* __restrict__ vals, const int* __restrict__ rows,
                  const int* __restrict__ cols, const float* __restrict__ X,
                  float* __restrict__ Y, int nE)
{
    int gid = blockIdx.x * 256 + threadIdx.x;
    int e = gid >> 5, lane = threadIdx.x & 31;
    if (e >= nE) return;
    float v = vals[e];
    const float* xp = X + (long)cols[e] * D;
    float*       yp = Y + (long)rows[e] * D;
    #pragma unroll
    for (int f4 = 0; f4 < D; f4 += 128) {
        int f = f4 + lane * 4;
        v4f xv = *(const v4f*)(xp + f);             // global_load_b128 gather
        #pragma unroll
        for (int j = 0; j < 4; ++j) {
            float t = xv[j]; if (SQ) t *= t;
            atomAddF(&yp[f + j], v * t);
        }
    }
}

__global__ void seg_deg(const float* vals, const int* rows, float* deg, int nE) {
    int e = blockIdx.x * blockDim.x + threadIdx.x;
    if (e < nE) atomAddF(&deg[rows[e]], vals[e]);
}

// per-column sums over a row chunk; blockDim.x == D (coalesced across cols)
__global__ void bn_stats(const float* __restrict__ X, int D, int nrows,
                         int rowsPerBlock, float* sums, float* sqs)
{
    int c = threadIdx.x;
    int r0 = blockIdx.x * rowsPerBlock;
    int r1 = r0 + rowsPerBlock; if (r1 > nrows) r1 = nrows;
    float s = 0.f, q = 0.f;
    for (int r = r0; r < r1; ++r) {
        float v = X[(long)r * D + c];
        s += v; q += v * v;
    }
    atomAddF(&sums[c], s);
    atomAddF(&sqs[c],  q);
}

// in-place batchnorm + relu (pre-BN bias cancels in (h-mu), so omitted)
__global__ void bn_apply_relu(float* __restrict__ X, const float* sums,
                              const float* sqs, const float* g, const float* b,
                              int D, int nrows)
{
    int i = blockIdx.x * blockDim.x + threadIdx.x;
    if (i >= nrows * D) return;
    int c = i % D;
    float inv = 1.f / (float)nrows;
    float mu  = sums[c] * inv;
    float var = sqs[c] * inv - mu * mu;
    float v = (X[i] - mu) * rsqrtf(var + 1e-5f) * g[c] + b[c];
    X[i] = fmaxf(v, 0.f);
}

__device__ __forceinline__ float block_sum_128(float v, float* red, int c) {
    red[c] = v; __syncthreads();
    for (int s = 64; s > 0; s >>= 1) {
        if (c < s) red[c] += red[c + s];
        __syncthreads();
    }
    float r = red[0]; __syncthreads();
    return r;
}

// in-place row layernorm (+optional relu), H=128, one block per row
__global__ __launch_bounds__(128)
void ln_128(float* __restrict__ X, const float* g, const float* b, int doRelu)
{
    __shared__ float red[128];
    int row = blockIdx.x, c = threadIdx.x;
    long i = (long)row * 128 + c;
    float v  = X[i];
    float mu = block_sum_128(v, red, c) * (1.f / 128.f);
    float d  = v - mu;
    float var = block_sum_128(d * d, red, c) * (1.f / 128.f);
    float o = d * rsqrtf(var + 1e-5f) * g[c] + b[c];
    X[i] = doRelu ? fmaxf(o, 0.f) : o;
}

// residual = LN(res_scale * (T2 - low)) in place of T2; T2 = x@Wxp + bxp
__global__ __launch_bounds__(128)
void resid_ln_128(float* __restrict__ T2, const float* __restrict__ low,
                  const float* rs, const float* g, const float* b)
{
    __shared__ float red[128];
    int row = blockIdx.x, c = threadIdx.x;
    long i = (long)row * 128 + c;
    float v  = rs[0] * (T2[i] - low[i]);
    float mu = block_sum_128(v, red, c) * (1.f / 128.f);
    float d  = v - mu;
    float var = block_sum_128(d * d, red, c) * (1.f / 128.f);
    T2[i] = d * rsqrtf(var + 1e-5f) * g[c] + b[c];
}

// x_cat[n] = [x, AX-x, x_std, deg, pad...], row stride CATLD
#define CATLD 776
__global__ void xcat_build(const float* __restrict__ x, const float* __restrict__ AX,
                           const float* __restrict__ AX2, const float* __restrict__ deg,
                           float* __restrict__ cat, int nrows)
{
    int i = blockIdx.x * blockDim.x + threadIdx.x;
    if (i >= nrows * 256) return;
    int n = i >> 8, c = i & 255;
    float xv = x[i], ax = AX[i], ax2 = AX2[i], dg = deg[n];
    float id = 1.f / (dg + 1e-8f);
    float m  = ax * id;
    float var = ax2 * id - m * m;
    float sd = sqrtf(fmaxf(var, 0.f));
    float* cp = cat + (long)n * CATLD;
    cp[c]        = xv;
    cp[256 + c]  = ax - xv;
    cp[512 + c]  = sd;
    if (c == 0) cp[768] = dg;
    if (c >= 1 && c < 8) cp[768 + c] = 0.f;   // zero pad cols 769..775
}

// gate = softmax((hg@Wg2 + bg2)/TEMP), one wave per node
__global__ __launch_bounds__(128)
void gate_kernel(const float* __restrict__ hg, const float* __restrict__ Wg2,
                 const float* __restrict__ bg2, float* __restrict__ gate, int nrows)
{
    int gid = blockIdx.x * 128 + threadIdx.x;
    int n = gid >> 5, lane = threadIdx.x & 31;
    if (n >= nrows) return;
    int f = lane * 4;
    v4f hv = *(const v4f*)(hg + (long)n * 128 + f);
    float z0 = 0.f, z1 = 0.f;
    #pragma unroll
    for (int j = 0; j < 4; ++j) {
        z0 += hv[j] * Wg2[(f + j) * 2 + 0];
        z1 += hv[j] * Wg2[(f + j) * 2 + 1];
    }
    for (int m = 16; m; m >>= 1) {
        z0 += __shfl_xor(z0, m, 32);
        z1 += __shfl_xor(z1, m, 32);
    }
    if (lane == 0) {
        z0 = (z0 + bg2[0]) * 0.5f;   // /TEMP=2
        z1 = (z1 + bg2[1]) * 0.5f;
        float mx = fmaxf(z0, z1);
        float e0 = __expf(z0 - mx), e1 = __expf(z1 - mx);
        float inv = 1.f / (e0 + e1);
        gate[(long)n * 2 + 0] = e0 * inv;
        gate[(long)n * 2 + 1] = e1 * inv;
    }
}

// exp_s[e] = exp(leaky(Q[row].K[col] / sqrt(H))); denom[row] += exp_s
__global__ __launch_bounds__(256)
void edge_score(const float* __restrict__ Q, const float* __restrict__ K,
                const int* __restrict__ rows, const int* __restrict__ cols,
                float* __restrict__ exps, float* __restrict__ denom, int nE)
{
    int gid = blockIdx.x * 256 + threadIdx.x;
    int e = gid >> 5, lane = threadIdx.x & 31;
    if (e >= nE) return;
    int r = rows[e], c = cols[e];
    v4f q = *(const v4f*)(Q + (long)r * 128 + lane * 4);
    v4f k = *(const v4f*)(K + (long)c * 128 + lane * 4);
    float s = q[0]*k[0] + q[1]*k[1] + q[2]*k[2] + q[3]*k[3];
    for (int m = 16; m; m >>= 1) s += __shfl_xor(s, m, 32);
    if (lane == 0) {
        s *= 0.08838834764831845f;              // 1/sqrt(128)
        float l = (s > 0.f) ? s : 0.2f * s;     // leaky relu, ALPHA=0.2
        float ev = __expf(l);
        exps[e] = ev;
        atomAddF(&denom[r], ev);
    }
}

// high[row] += (exp_s / denom[row]) * V[col]
__global__ __launch_bounds__(256)
void attn_agg(const float* __restrict__ exps, const float* __restrict__ denom,
              const float* __restrict__ V, const int* __restrict__ rows,
              const int* __restrict__ cols, float* __restrict__ high, int nE)
{
    int gid = blockIdx.x * 256 + threadIdx.x;
    int e = gid >> 5, lane = threadIdx.x & 31;
    if (e >= nE) return;
    int r = rows[e], c = cols[e];
    float a = exps[e] / (denom[r] + 1e-16f);
    int f = lane * 4;
    v4f vv = *(const v4f*)(V + (long)c * 128 + f);
    float* hp = high + (long)r * 128 + f;
    #pragma unroll
    for (int j = 0; j < 4; ++j) atomAddF(&hp[j], a * vv[j]);
}

__global__ void final_out(const float* __restrict__ low, const float* __restrict__ high,
                          const float* __restrict__ gate, float* __restrict__ out, int nrows)
{
    int i = blockIdx.x * blockDim.x + threadIdx.x;
    if (i >= nrows * 128) return;
    int n = i >> 7;
    out[i] = gate[(long)n * 2 + 0] * low[i] + gate[(long)n * 2 + 1] * high[i];
}

// ---------------------------------------------------------------------------
// Launcher
// ---------------------------------------------------------------------------
extern "C" void kernel_launch(void* const* d_in, const int* in_sizes, int n_in,
                              void* d_out, int out_size, void* d_ws, size_t ws_size,
                              hipStream_t stream)
{
    const int N = 100000, E = 3200000, F0 = 256, H = 128, H2 = 256, CATD = 769;

    const float* x      = (const float*)d_in[0];
    const int*   erow   = (const int*)d_in[1];
    const int*   ecol   = (const int*)d_in[2];
    const float* avals  = (const float*)d_in[3];
    const float* W_gc1  = (const float*)d_in[4];
    /* b_gc1 (d_in[5]) cancels inside batchnorm mean subtraction */
    const float* bn1_g  = (const float*)d_in[6];
    const float* bn1_b  = (const float*)d_in[7];
    const float* W_gc2  = (const float*)d_in[8];
    /* b_gc2 (d_in[9]) cancels likewise */
    const float* bn2_g  = (const float*)d_in[10];
    const float* bn2_b  = (const float*)d_in[11];
    const float* Wg1    = (const float*)d_in[12];
    const float* bg1    = (const float*)d_in[13];
    const float* ln_g   = (const float*)d_in[14];
    const float* ln_b   = (const float*)d_in[15];
    const float* Wg2    = (const float*)d_in[16];
    const float* bg2    = (const float*)d_in[17];
    const float* Wxp    = (const float*)d_in[18];
    const float* bxp    = (const float*)d_in[19];
    const float* rscale = (const float*)d_in[20];
    const float* rn_g   = (const float*)d_in[21];
    const float* rn_b   = (const float*)d_in[22];
    const float* Wq     = (const float*)d_in[23];
    const float* Wk     = (const float*)d_in[24];
    const float* Wv     = (const float*)d_in[25];

    // d_out layout: output [N,128], gate [N,2], low [N,128], high [N,128]
    float* out  = (float*)d_out;
    float* gate = out  + (size_t)N * H;
    float* low  = gate + (size_t)N * 2;
    float* high = low  + (size_t)N * H;

    // workspace carve-up (f32)
    float* ws = (float*)d_ws;
    size_t off = 0;
    auto carve = [&](size_t n) { float* p = ws + off; off += (n + 3) & ~(size_t)3; return p; };
    float* bufA   = carve((size_t)N * 256);   // t0 = x@W_gc1 ; later AX
    float* bufB   = carve((size_t)N * 256);   // s0/h1        ; later AX2
    float* bufC   = carve((size_t)N * 128);   // t1 = h1@W_gc2; later residual
    float* bufQ   = carve((size_t)N * 128);
    float* bufK   = carve((size_t)N * 128);
    float* bufV   = carve((size_t)N * 128);
    float* bufHG  = carve((size_t)N * 128);
    float* bufCat = carve((size_t)N * CATLD);
    float* deg    = carve(N);
    float* exps   = carve(E);
    float* denom  = carve(N);
    float* csum   = carve(256);
    float* csq    = carve(256);

    auto zgrid = [](size_t n) { return dim3((unsigned)((n + 255) / 256)); };
    auto ggrid = [](int nc, int nr) { return dim3((nc + BN - 1) / BN, (nr + BM - 1) / BM); };
    const int edgeBlocks = (E + 7) / 8;            // one wave32 per edge
    const int rp = 128, statBlocks = (N + rp - 1) / rp;

    // ---- LowPass: t0 = x @ W_gc1
    gemm_bf16_wmma<<<ggrid(H2, N), 256, 0, stream>>>(x, F0, W_gc1, H2, nullptr, bufA, H2, N, F0, H2);
    // s0 = spmm(adj, t0)
    zero_f32<<<zgrid((size_t)N * 256), 256, 0, stream>>>(bufB, (size_t)N * 256);
    spmm_scatter<256,false><<<edgeBlocks, 256, 0, stream>>>(avals, erow, ecol, bufA, bufB, E);
    // h1 = relu(bn1(s0))
    zero_f32<<<zgrid(256), 256, 0, stream>>>(csum, 256);
    zero_f32<<<zgrid(256), 256, 0, stream>>>(csq, 256);
    bn_stats<<<statBlocks, 256, 0, stream>>>(bufB, 256, N, rp, csum, csq);
    bn_apply_relu<<<zgrid((size_t)N * 256), 256, 0, stream>>>(bufB, csum, csq, bn1_g, bn1_b, 256, N);
    // t1 = h1 @ W_gc2
    gemm_bf16_wmma<<<ggrid(H, N), 256, 0, stream>>>(bufB, H2, W_gc2, H, nullptr, bufC, H, N, H2, H);
    // low = relu(bn2(spmm(adj, t1)))
    zero_f32<<<zgrid((size_t)N * 128), 256, 0, stream>>>(low, (size_t)N * 128);
    spmm_scatter<128,false><<<edgeBlocks, 256, 0, stream>>>(avals, erow, ecol, bufC, low, E);
    zero_f32<<<zgrid(256), 256, 0, stream>>>(csum, 256);
    zero_f32<<<zgrid(256), 256, 0, stream>>>(csq, 256);
    bn_stats<<<statBlocks, 128, 0, stream>>>(low, 128, N, rp, csum, csq);
    bn_apply_relu<<<zgrid((size_t)N * 128), 256, 0, stream>>>(low, csum, csq, bn2_g, bn2_b, 128, N);

    // ---- Gating stats: AX, AX2, deg
    zero_f32<<<zgrid((size_t)N * 256), 256, 0, stream>>>(bufA, (size_t)N * 256);
    spmm_scatter<256,false><<<edgeBlocks, 256, 0, stream>>>(avals, erow, ecol, x, bufA, E);
    zero_f32<<<zgrid((size_t)N * 256), 256, 0, stream>>>(bufB, (size_t)N * 256);
    spmm_scatter<256,true><<<edgeBlocks, 256, 0, stream>>>(avals, erow, ecol, x, bufB, E);
    zero_f32<<<zgrid(N), 256, 0, stream>>>(deg, N);
    seg_deg<<<(E + 255) / 256, 256, 0, stream>>>(avals, erow, deg, E);
    xcat_build<<<zgrid((size_t)N * 256), 256, 0, stream>>>(x, bufA, bufB, deg, bufCat, N);
    // hg = relu(LN(x_cat @ Wg1 + bg1))
    gemm_bf16_wmma<<<ggrid(H, N), 256, 0, stream>>>(bufCat, CATLD, Wg1, H, bg1, bufHG, H, N, CATD, H);
    ln_128<<<N, 128, 0, stream>>>(bufHG, ln_g, ln_b, 1);
    gate_kernel<<<(N * 32 + 127) / 128, 128, 0, stream>>>(bufHG, Wg2, bg2, gate, N);

    // ---- HighPass: residual = LN(rs*(x@Wxp + bxp - low))
    gemm_bf16_wmma<<<ggrid(H, N), 256, 0, stream>>>(x, F0, Wxp, H, bxp, bufC, H, N, F0, H);
    resid_ln_128<<<N, 128, 0, stream>>>(bufC, low, rscale, rn_g, rn_b);
    // Q, K, V
    gemm_bf16_wmma<<<ggrid(H, N), 256, 0, stream>>>(bufC, H, Wq, H, nullptr, bufQ, H, N, H, H);
    gemm_bf16_wmma<<<ggrid(H, N), 256, 0, stream>>>(bufC, H, Wk, H, nullptr, bufK, H, N, H, H);
    gemm_bf16_wmma<<<ggrid(H, N), 256, 0, stream>>>(bufC, H, Wv, H, nullptr, bufV, H, N, H, H);
    // sparse attention
    zero_f32<<<zgrid(N), 256, 0, stream>>>(denom, N);
    edge_score<<<edgeBlocks, 256, 0, stream>>>(bufQ, bufK, erow, ecol, exps, denom, E);
    zero_f32<<<zgrid((size_t)N * 128), 256, 0, stream>>>(high, (size_t)N * 128);
    attn_agg<<<edgeBlocks, 256, 0, stream>>>(exps, denom, bufV, erow, ecol, high, E);

    // ---- mix
    final_out<<<zgrid((size_t)N * 128), 256, 0, stream>>>(low, high, gate, out, N);

    (void)in_sizes; (void)n_in; (void)out_size; (void)ws_size;
}
